// ShiftLoss_15204184228313
// MI455X (gfx1250) — matrix operand order repaired and checked by reading, not compiled, not run on hardware
//
#include <hip/hip_runtime.h>
#include <math.h>

typedef __attribute__((ext_vector_type(16))) _Float16 v16h;
typedef __attribute__((ext_vector_type(4)))  _Float16 v4h;
typedef __attribute__((ext_vector_type(8)))  float    v8f;

#define T_LEN   4096
#define SHIFT   10
#define NSHIFT  (2*SHIFT+1)
#define PAD     16
#define H_LEN   4160                      // zero-padded f16 length (>= T+48)
#define BLOCK   128
#define NWAVE   (BLOCK/32)
#define NCHUNK  ((T_LEN + 2*PAD)/32)      // 129 chunks of K=32

__global__ void zero_out(float* out) { out[0] = 0.f; }

__global__ __launch_bounds__(BLOCK)
void shift_loss_kernel(const float* __restrict__ x,
                       const float* __restrict__ y,
                       float* __restrict__ out, int B)
{
    __shared__ __align__(16) _Float16 xH[H_LEN];   // zero-padded f16 row copies
    __shared__ __align__(16) _Float16 yH[H_LEN];
    __shared__ __align__(16) float    xF[T_LEN];   // exact fp32 row copies
    __shared__ __align__(16) float    yF[T_LEN];
    __shared__ float    Cws[NWAVE][16][16];
    __shared__ float4   red4[BLOCK];
    __shared__ float    red [BLOCK];
    __shared__ float    sxy16[NSHIFT];
    __shared__ float    fxx[SHIFT+1], bxx[SHIFT+1], fyy[SHIFT+1], byy[SHIFT+1];
    __shared__ int      sBest;

    const int row  = blockIdx.x;
    const int tid  = threadIdx.x;
    const int lane = tid & 31;
    const int wave = tid >> 5;
    const float4* xr4 = (const float4*)(x + (size_t)row * T_LEN);
    const float4* yr4 = (const float4*)(y + (size_t)row * T_LEN);

    // ---- Load row once (float4-wide): fp32 LDS copies, f16 copies, fused totals ----
    float sx = 0.f, sxx = 0.f, sy = 0.f, syy = 0.f;
    {
        float4* xF4 = (float4*)xF;
        float4* yF4 = (float4*)yF;
        for (int v = tid; v < T_LEN/4; v += BLOCK) {
            const float4 xv = xr4[v];
            const float4 yv = yr4[v];
            xF4[v] = xv;  yF4[v] = yv;
            sx  += xv.x + xv.y + xv.z + xv.w;
            sxx += xv.x*xv.x + xv.y*xv.y + xv.z*xv.z + xv.w*xv.w;
            sy  += yv.x + yv.y + yv.z + yv.w;
            syy += yv.x*yv.x + yv.y*yv.y + yv.z*yv.z + yv.w*yv.w;
            v4h xh = { (_Float16)xv.x, (_Float16)xv.y, (_Float16)xv.z, (_Float16)xv.w };
            v4h yh = { (_Float16)yv.x, (_Float16)yv.y, (_Float16)yv.z, (_Float16)yv.w };
            *(v4h*)&xH[PAD + 4*v] = xh;      // 8-byte aligned
            *(v4h*)&yH[PAD + 4*v] = yh;
        }
        for (int i = tid; i < PAD; i += BLOCK) {           // front pad
            xH[i] = (_Float16)0.f;  yH[i] = (_Float16)0.f;
        }
        for (int i = PAD + T_LEN + tid; i < H_LEN; i += BLOCK) {  // back pad
            xH[i] = (_Float16)0.f;  yH[i] = (_Float16)0.f;
        }
    }
    red4[tid] = make_float4(sx, sxx, sy, syy);
    __syncthreads();
    for (int o = BLOCK/2; o > 0; o >>= 1) {
        if (tid < o) {
            red4[tid].x += red4[tid+o].x;  red4[tid].y += red4[tid+o].y;
            red4[tid].z += red4[tid+o].z;  red4[tid].w += red4[tid+o].w;
        }
        __syncthreads();
    }

    // ---- WMMA Toeplitz cross-correlation: D[m,n] accumulates Sxy(m+n-15) ----
    // A[m,k] = yH[j0+k+m]  (16x32 f16),  B[k,n] = xH[j0+k+15-n]  (32x16 f16)
    // A lane layout: M=lane%16; halves 0..7 -> K = 8*(lane>=16)+h ; halves 8..15 -> K+16
    // B lane layout: N=lane%16; halves 0..7 -> K = 8*(lane>=16)+h ; halves 8..15 -> K+16
    const int M  = lane & 15;
    const int hi = lane >> 4;
    const int aO = M + 8*hi;
    const int bO = (15 - M) + 8*hi;

    auto loadFrag = [&](int j0, v16h& a, v16h& b) {
#pragma unroll
        for (int h = 0; h < 8; ++h) {
            a[h]     = yH[j0 + aO + h];
            a[h + 8] = yH[j0 + aO + 16 + h];
        }
#pragma unroll
        for (int h = 0; h < 8; ++h) {
            b[h]     = xH[j0 + bO + h];
            b[h + 8] = xH[j0 + bO + 16 + h];
        }
    };

    v8f acc0 = {}, acc1 = {};
    int c = wave;
    // unroll x2 with independent accumulators: loads of chunk c+NWAVE overlap WMMA of c
    for (; c + NWAVE < NCHUNK; c += 2*NWAVE) {
        v16h a0, b0, a1, b1;
        loadFrag(c * 32,           a0, b0);
        loadFrag((c + NWAVE) * 32, a1, b1);
        acc0 = __builtin_amdgcn_wmma_f32_16x16x32_f16(false, a0, false, b0,
                                                      (short)0, acc0, false, false);
        acc1 = __builtin_amdgcn_wmma_f32_16x16x32_f16(false, a1, false, b1,
                                                      (short)0, acc1, false, false);
    }
    if (c < NCHUNK) {
        v16h a0, b0;
        loadFrag(c * 32, a0, b0);
        acc0 = __builtin_amdgcn_wmma_f32_16x16x32_f16(false, a0, false, b0,
                                                      (short)0, acc0, false, false);
    }
    acc0 = acc0 + acc1;

    // C/D layout: lanes 0-15: N=lane, M=r ; lanes 16-31: N=lane-16, M=r+8
#pragma unroll
    for (int r = 0; r < 8; ++r)
        Cws[wave][hi ? r + 8 : r][M] = acc0[r];
    __syncthreads();

    if (tid < NSHIFT) {                       // s = tid-10, sigma15 = s+15
        const int sig = tid + 5;
        const int m = sig > 15 ? 15 : sig;
        const int n = sig - m;
        float v = 0.f;
        for (int w = 0; w < NWAVE; ++w) v += Cws[w][m][n];
        sxy16[tid] = v;
    }
    __syncthreads();

    // ---- Argmax Pearson over 21 shifts (fp32 sums + f16-derived Sxy) ----
    if (tid == 0) {
        const float Sx = red4[0].x, Sxx = red4[0].y, Sy = red4[0].z, Syy = red4[0].w;
        float fx[SHIFT+1], bx[SHIFT+1], fy[SHIFT+1], by[SHIFT+1];
        fx[0]=bx[0]=fy[0]=by[0]=0.f; fxx[0]=bxx[0]=fyy[0]=byy[0]=0.f;
        for (int k = 1; k <= SHIFT; ++k) {
            const float xf = xF[k-1],       xb = xF[T_LEN-k];
            const float yf = yF[k-1],       yb = yF[T_LEN-k];
            fx[k]=fx[k-1]+xf;   bx[k]=bx[k-1]+xb;
            fy[k]=fy[k-1]+yf;   by[k]=by[k-1]+yb;
            fxx[k]=fxx[k-1]+xf*xf; bxx[k]=bxx[k-1]+xb*xb;
            fyy[k]=fyy[k-1]+yf*yf; byy[k]=byy[k-1]+yb*yb;
        }
        float best = -INFINITY; int bs = 0;
        for (int s = -SHIFT; s <= SHIFT; ++s) {
            const int a   = s >= 0 ? s : -s;
            const float n = (float)(T_LEN - a);
            const float Sxw  = Sx  - (s >= 0 ? bx [a] : fx [a]);
            const float Sxxw = Sxx - (s >= 0 ? bxx[a] : fxx[a]);
            const float Syw  = Sy  - (s >= 0 ? fy [a] : by [a]);
            const float Syyw = Syy - (s >= 0 ? fyy[a] : byy[a]);
            const float mx = Sxw / n, my = Syw / n;
            const float cov = sxy16[s + SHIFT] - n * mx * my;
            const float vx  = Sxxw - n * mx * mx;
            const float vy  = Syyw - n * my * my;
            const float corr = cov / sqrtf(vx * vy);
            if (corr > best) { best = corr; bs = s; }
        }
        sBest = bs;
    }
    __syncthreads();

    // ---- Precise fp32 Sxy at best shift from LDS copies ----
    const int s   = sBest;
    const int lo  = s < 0 ? -s : 0;
    const int hiJ = s > 0 ? T_LEN - s : T_LEN;
    float p = 0.f;
    for (int j = lo + tid; j < hiJ; j += BLOCK) p += xF[j] * yF[j + s];
    red[tid] = p;
    __syncthreads();
    for (int o = BLOCK/2; o > 0; o >>= 1) {
        if (tid < o) red[tid] += red[tid + o];
        __syncthreads();
    }
    if (tid == 0) {
        const int a = s >= 0 ? s : -s;
        const float n    = (float)(T_LEN - a);
        const float Sxxw = red4[0].y - (s >= 0 ? bxx[a] : fxx[a]);
        const float Syyw = red4[0].w - (s >= 0 ? fyy[a] : byy[a]);
        const float mse  = (Sxxw + Syyw - 2.f * red[0]) / n;
        atomicAdd(out, mse / (float)B);
    }
}

extern "C" void kernel_launch(void* const* d_in, const int* in_sizes, int n_in,
                              void* d_out, int out_size, void* d_ws, size_t ws_size,
                              hipStream_t stream) {
    const float* x = (const float*)d_in[0];
    const float* y = (const float*)d_in[1];
    float* out = (float*)d_out;
    const int B = in_sizes[0] / T_LEN;
    hipLaunchKernelGGL(zero_out, dim3(1), dim3(1), 0, stream, out);
    hipLaunchKernelGGL(shift_loss_kernel, dim3(B), dim3(BLOCK), 0, stream,
                       x, y, out, B);
}
